// Seq2Seq_38560216383693
// MI455X (gfx1250) — matrix operand order repaired and checked by reading, not compiled, not run on hardware
//
#include <hip/hip_runtime.h>
#include <hip/hip_bf16.h>
#include <stdint.h>

typedef unsigned int u32;
typedef unsigned short u16;
typedef unsigned long long u64;

#define BB 16
#define SS 64
#define TT 64
#define VV 32000
#define EE 512
#define HH 1024
#define HDD 512

#define APITCH 1544   // bf16 elems per LDS row (1536+8 pad): 3088B row, 16B aligned, bank-skewed
#define GPITCH 1032   // generator LDS pitch (1024+8 pad): 2064B row

typedef __attribute__((ext_vector_type(16))) __bf16 v16bf;
typedef __attribute__((ext_vector_type(8)))  float  v8f;

union Frag16 { v16bf v; u32 u[8]; uint4 q[2]; };
union FragC  { v8f  v; float f[8]; };

__device__ __forceinline__ u16 f2bf(float x) {
  union { float f; u32 u; } c; c.f = x;
  u32 r = c.u + 0x7FFFu + ((c.u >> 16) & 1u);   // RNE
  return (u16)(r >> 16);
}
__device__ __forceinline__ float bf2f(u16 h) {
  union { u32 u; float f; } c; c.u = ((u32)h) << 16; return c.f;
}
__device__ __forceinline__ u32 pk2(float a, float b) {
  return (u32)f2bf(a) | ((u32)f2bf(b) << 16);
}
__device__ __forceinline__ float rcp_f(float x) { return __builtin_amdgcn_rcpf(x); }
// branch-free activations (avoid ocml tanhf's divergent range splits)
__device__ __forceinline__ float sigm(float x) { return rcp_f(1.0f + __expf(-x)); }
__device__ __forceinline__ float tanh_f(float x) { return 1.0f - 2.0f * rcp_f(1.0f + __expf(x + x)); }

__device__ __forceinline__ float wsum(float x) {
#pragma unroll
  for (int m = 16; m; m >>= 1) x += __shfl_xor(x, m, 32);
  return x;
}
__device__ __forceinline__ float wmaxr(float x) {
#pragma unroll
  for (int m = 16; m; m >>= 1) x = fmaxf(x, __shfl_xor(x, m, 32));
  return x;
}

// ---- CDNA5 async global->LDS copy (ASYNCcnt-tracked, no VGPR round trip) ----
__device__ __forceinline__ u32 lds_addr(const void* p) {
  // LDS aperture keeps the wave-relative LDS byte offset in addr[31:0]
  return (u32)(size_t)p;
}
__device__ __forceinline__ void async_cp16(u32 ldsByte, u64 gaddr) {
  asm volatile("global_load_async_to_lds_b128 %0, %1, off"
               :: "v"(ldsByte), "v"(gaddr) : "memory");
}
__device__ __forceinline__ void async_wait0() {
  asm volatile("s_wait_asynccnt 0" ::: "memory");
}

// A fragment (16x32 bf16, M=batch) from LDS row-major staging.
// lanes 0-15: K {0..15}; lanes 16-31: K {16..31}; M = lane&15.
__device__ __forceinline__ void loadA(Frag16& a, const u16* lds, int pitchE, int kt) {
  int lane = threadIdx.x & 31;
  const uint4* row = (const uint4*)(lds + (lane & 15) * pitchE);
  int b0 = kt * 4 + ((lane >> 4) << 1);
  a.q[0] = row[b0];
  a.q[1] = row[b0 + 1];
}
// B fragment (32x16 bf16) from packed tile: lane-major, 32B/lane contiguous.
__device__ __forceinline__ void loadB(Frag16& b, const u32* tileBase) {
  int lane = threadIdx.x & 31;
  const uint4* p = (const uint4*)(tileBase + lane * 8);
  b.q[0] = p[0];
  b.q[1] = p[1];
}
__device__ __forceinline__ v8f wmma_bf(const Frag16& a, const Frag16& b, v8f c) {
  return __builtin_amdgcn_wmma_f32_16x16x32_bf16(false, a.v, false, b.v, (short)0, c, false, false);
}

// 4x4-tile gate GEMM step: weight base pre-offset by wave ((w*4)*KT*256);
// per-k one pointer bump, 16 B-loads via immediate offsets, 16 WMMA.
template<int KT>
__device__ __forceinline__ void gemm_gates4(FragC (&acc)[4][4], const u16* lds,
                                            const u32* wbW, int ktA0) {
  for (int kt = 0; kt < KT; ++kt) {
    Frag16 a; loadA(a, lds, APITCH, ktA0 + kt);
    const u32* pk0 = wbW + (size_t)kt * 256;
#pragma unroll
    for (int r = 0; r < 4; ++r) {
      const u32* pr = pk0 + (size_t)r * (64 * KT * 256);
#pragma unroll
      for (int q = 0; q < 4; ++q) {
        Frag16 bfr; loadB(bfr, pr + (size_t)q * (KT * 256));
        acc[r][q].v = wmma_bf(a, bfr, acc[r][q].v);
      }
    }
  }
}
// 4-tile row GEMM step (q/attn/concat): 32 k-frags per chunk.
template<int KT>
__device__ __forceinline__ void gemm_row4(FragC (&acc)[4], const u16* lds,
                                          const u32* wbW, int ktA0, int ktW0) {
  for (int kt = 0; kt < 32; ++kt) {
    Frag16 a; loadA(a, lds, APITCH, ktA0 + kt);
    const u32* pk0 = wbW + (size_t)(ktW0 + kt) * 256;
#pragma unroll
    for (int q = 0; q < 4; ++q) {
      Frag16 bfr; loadB(bfr, pk0 + (size_t)q * (KT * 256));
      acc[q].v = wmma_bf(a, bfr, acc[q].v);
    }
  }
}
// encoder 4x2-tile gate GEMM; base pre-offset by (w*2)*KT*256.
template<int KT>
__device__ __forceinline__ void gemm_enc(FragC (&acc)[4][2], const u16* lds,
                                         const u32* wbW, int ktA0) {
  for (int kt = 0; kt < KT; ++kt) {
    Frag16 a; loadA(a, lds, APITCH, ktA0 + kt);
    const u32* pk0 = wbW + (size_t)kt * 256;
#pragma unroll
    for (int r = 0; r < 4; ++r) {
      const u32* pr = pk0 + (size_t)r * (32 * KT * 256);
#pragma unroll
      for (int g = 0; g < 2; ++g) {
        Frag16 bfr; loadB(bfr, pr + (size_t)g * (KT * 256));
        acc[r][g].v = wmma_bf(a, bfr, acc[r][g].v);
      }
    }
  }
}

// ---------------- weight packing: (N x K) f32 -> bf16 B-fragment tiles ----------------
__global__ void k_pack(const float* __restrict__ W, u32* __restrict__ P, int N, int K) {
  int kT = K >> 5;
  int tile = blockIdx.x;
  int nt = tile / kT, kt = tile - nt * kT;
  int lane = threadIdx.x;
  int n = nt * 16 + (lane & 15);
  int kbase = kt * 32 + ((lane >> 4) << 4);
  const float* wr = W + (size_t)n * K + kbase;
  u32* out = P + (size_t)tile * 256 + lane * 8;
#pragma unroll
  for (int j = 0; j < 8; ++j) out[j] = pk2(wr[2 * j], wr[2 * j + 1]);
}

__global__ void k_bias(const float* __restrict__ a, const float* __restrict__ b,
                       float* __restrict__ o, int n) {
  int i = blockIdx.x * 256 + threadIdx.x;
  if (i < n) o[i] = a[i] + b[i];
}

// embedding gather -> bf16 (t, b, e)
__global__ void k_gather(const float* __restrict__ tab, const int* __restrict__ ids,
                         u16* __restrict__ out, int seqLen) {
  int t = blockIdx.x >> 4, b = blockIdx.x & 15;
  int id = ids[b * seqLen + t];
  const float* row = tab + (size_t)id * EE;
  u16* o = out + ((size_t)t * BB + b) * EE;
  for (int e = threadIdx.x; e < EE; e += blockDim.x) o[e] = f2bf(row[e]);
}

// ---------------- encoder layer: 2 blocks (fwd/bwd), 16 waves, in-kernel time loop ----------------
__global__ __launch_bounds__(512) void k_enc(
    const u16* __restrict__ xs, int K1,
    const u32* __restrict__ ihF, const u32* __restrict__ hhF, const float* __restrict__ bF,
    const u32* __restrict__ ihB, const u32* __restrict__ hhB, const float* __restrict__ bB,
    u16* __restrict__ yOut, int hsrcLayout, float* __restrict__ hcFin) {
  __shared__ __align__(16) u16 bufA[BB * APITCH];
  int dir = blockIdx.x;
  const u32* ih = dir ? ihB : ihF;
  const u32* hh = dir ? hhB : hhF;
  const float* bias = dir ? bB : bF;
  int tid = threadIdx.x;
  int w = tid >> 5, lane = tid & 31, hi = lane >> 4, l15 = lane & 15;
  int kT1 = K1 >> 5;
  int cpr = K1 >> 3;                     // 16B chunks per staged row
  u32 ldsBase = lds_addr(bufA);

  for (int i = tid; i < BB * HDD; i += 512) {   // zero h region
    int b = i >> 9, j = i & 511;
    bufA[b * APITCH + K1 + j] = (u16)0;
  }
  float cst[2][8];
#pragma unroll
  for (int g = 0; g < 2; ++g)
#pragma unroll
    for (int v = 0; v < 8; ++v) cst[g][v] = 0.0f;
  __syncthreads();

  const u32* ihW = ih + (size_t)(w * 2) * kT1 * 256;
  const u32* hhW = hh + (size_t)(w * 2) * 16 * 256;

  for (int step = 0; step < SS; ++step) {
    int t = dir ? (SS - 1 - step) : step;
    {  // async-stage x_t into cols [0, K1)
      u64 src = (u64)(size_t)(xs + (size_t)t * BB * K1);
      for (int i = tid; i < BB * cpr; i += 512) {
        int b = i / cpr, c = i - b * cpr;
        async_cp16(ldsBase + (u32)(b * (APITCH * 2) + c * 16),
                   src + (u64)b * (K1 * 2) + (u64)c * 16);
      }
      async_wait0();
    }
    __syncthreads();

    FragC acc[4][2];
#pragma unroll
    for (int r = 0; r < 4; ++r)
#pragma unroll
      for (int g = 0; g < 2; ++g) {
        float bv = bias[r * HDD + w * 32 + g * 16 + l15];
#pragma unroll
        for (int v = 0; v < 8; ++v) acc[r][g].f[v] = bv;
      }
    if (kT1 == 16) gemm_enc<16>(acc, bufA, ihW, 0);
    else           gemm_enc<32>(acc, bufA, ihW, 0);
    gemm_enc<16>(acc, bufA, hhW, kT1);

    // LSTM cell (i,f,g,o)
    u16 hbf[2][8]; float hfv[2][8], cfv[2][8];
#pragma unroll
    for (int g = 0; g < 2; ++g)
#pragma unroll
      for (int v = 0; v < 8; ++v) {
        float iv = sigm(acc[0][g].f[v]);
        float fv = sigm(acc[1][g].f[v]);
        float gv = tanh_f(acc[2][g].f[v]);
        float ov = sigm(acc[3][g].f[v]);
        float c = fv * cst[g][v] + iv * gv;
        cst[g][v] = c;
        float h = ov * tanh_f(c);
        hfv[g][v] = h; cfv[g][v] = c; hbf[g][v] = f2bf(h);
      }
    __syncthreads();
#pragma unroll
    for (int g = 0; g < 2; ++g)
#pragma unroll
      for (int v = 0; v < 8; ++v) {
        int b = v + (hi << 3);
        int j = w * 32 + g * 16 + l15;
        bufA[b * APITCH + K1 + j] = hbf[g][v];
        size_t oi = hsrcLayout ? ((size_t)b * (SS * HH) + (size_t)t * HH + dir * HDD + j)
                               : ((size_t)t * (BB * HH) + (size_t)b * HH + dir * HDD + j);
        yOut[oi] = hbf[g][v];
        if (step == SS - 1) {
          hcFin[((size_t)(dir * 2 + 0) * BB + b) * HDD + j] = hfv[g][v];
          hcFin[((size_t)(dir * 2 + 1) * BB + b) * HDD + j] = cfv[g][v];
        }
      }
    __syncthreads();
  }
}

__global__ void k_decinit(const float* __restrict__ hc0, const float* __restrict__ hc1,
                          u16* __restrict__ h_state, float* __restrict__ c_state,
                          u16* __restrict__ htilde) {
  int i = blockIdx.x * 256 + threadIdx.x;
  if (i < 2 * BB * HH) {
    int l = i >> 14;
    int b = (i >> 10) & 15;
    int j = i & 1023;
    int dir = j >> 9, jj = j & 511;
    const float* hc = l ? hc1 : hc0;
    h_state[i] = f2bf(hc[((size_t)(dir * 2 + 0) * BB + b) * HDD + jj]);
    c_state[i] =      hc[((size_t)(dir * 2 + 1) * BB + b) * HDD + jj];
  }
  if (i < BB * HH) htilde[i] = (u16)0;
}

// ---------------- decoder: single block, 16 waves, 64 sequential steps ----------------
__global__ __launch_bounds__(512) void k_dec(
    const u16* __restrict__ embt, const u16* __restrict__ hsrc,
    u16* __restrict__ h_state, float* __restrict__ c_state,
    u16* __restrict__ htilde_state, float* __restrict__ q_buf, u16* __restrict__ htl_all,
    const u32* __restrict__ d0ih, const u32* __restrict__ d0hh, const float* __restrict__ bias0,
    const u32* __restrict__ d1ih, const u32* __restrict__ d1hh, const float* __restrict__ bias1,
    const u32* __restrict__ attn, const u32* __restrict__ ccw, const float* __restrict__ ccb) {
  __shared__ __align__(16) u16 bufA[BB * APITCH];
  __shared__ float scoreB[BB * SS];
  int tid = threadIdx.x;
  int w = tid >> 5, lane = tid & 31, hi = lane >> 4, l15 = lane & 15;
  u32 ldsBase = lds_addr(bufA);

  const u32* d0ihW = d0ih + (size_t)(w * 4) * 48 * 256;
  const u32* d0hhW = d0hh + (size_t)(w * 4) * 32 * 256;
  const u32* d1ihW = d1ih + (size_t)(w * 4) * 32 * 256;
  const u32* d1hhW = d1hh + (size_t)(w * 4) * 32 * 256;
  const u32* attnW = attn + (size_t)(w * 4) * 32 * 256;
  const u32* ccwW  = ccw  + (size_t)(w * 4) * 64 * 256;
  u64 gHst = (u64)(size_t)h_state;
  u64 gHtl = (u64)(size_t)htilde_state;

  for (int t = 0; t < TT; ++t) {
    {  // async-stage [emb_t (512) | h_tilde (1024)] : 192 x 16B chunks per row
      u64 gEmb = (u64)(size_t)(embt + (size_t)t * BB * EE);
      for (int i = tid; i < BB * 192; i += 512) {
        int b = i / 192, c = i - b * 192;
        u64 src = (c < 64) ? (gEmb + (u64)b * 1024 + (u64)c * 16)
                           : (gHtl + (u64)b * 2048 + (u64)(c - 64) * 16);
        async_cp16(ldsBase + (u32)(b * (APITCH * 2) + c * 16), src);
      }
      async_wait0();
    }
    __syncthreads();

    FragC acc[4][4];
    // ---- layer0 gates: K = 1536 (x) + 1024 (h0) ----
#pragma unroll
    for (int r = 0; r < 4; ++r)
#pragma unroll
      for (int q = 0; q < 4; ++q) {
        float bv = bias0[r * HH + w * 64 + q * 16 + l15];
#pragma unroll
        for (int v = 0; v < 8; ++v) acc[r][q].f[v] = bv;
      }
    gemm_gates4<48>(acc, bufA, d0ihW, 0);
    __syncthreads();
    for (int i = tid; i < BB * 128; i += 512) {   // stage h0_state (128 chunks/row)
      int b = i >> 7, c = i & 127;
      async_cp16(ldsBase + (u32)(b * (APITCH * 2) + c * 16), gHst + (u64)b * 2048 + (u64)c * 16);
    }
    async_wait0();
    __syncthreads();
    gemm_gates4<32>(acc, bufA, d0hhW, 0);

    u16 hb0[4][8];
#pragma unroll
    for (int q = 0; q < 4; ++q)
#pragma unroll
      for (int v = 0; v < 8; ++v) {
        int b = v + (hi << 3);
        int j = w * 64 + q * 16 + l15;
        float iv = sigm(acc[0][q].f[v]);
        float fv = sigm(acc[1][q].f[v]);
        float gv = tanh_f(acc[2][q].f[v]);
        float ov = sigm(acc[3][q].f[v]);
        size_t ci = (size_t)b * HH + j;
        float c = fv * c_state[ci] + iv * gv;
        c_state[ci] = c;
        hb0[q][v] = f2bf(ov * tanh_f(c));
      }
    __syncthreads();
#pragma unroll
    for (int q = 0; q < 4; ++q)
#pragma unroll
      for (int v = 0; v < 8; ++v) {
        int b = v + (hi << 3);
        int j = w * 64 + q * 16 + l15;
        bufA[b * APITCH + j] = hb0[q][v];
        h_state[(size_t)b * HH + j] = hb0[q][v];
      }
    __syncthreads();

    // ---- layer1 gates: K = 1024 (h_l0) + 1024 (h1) ----
#pragma unroll
    for (int r = 0; r < 4; ++r)
#pragma unroll
      for (int q = 0; q < 4; ++q) {
        float bv = bias1[r * HH + w * 64 + q * 16 + l15];
#pragma unroll
        for (int v = 0; v < 8; ++v) acc[r][q].f[v] = bv;
      }
    gemm_gates4<32>(acc, bufA, d1ihW, 0);
    __syncthreads();
    for (int i = tid; i < BB * 128; i += 512) {   // stage h1_state
      int b = i >> 7, c = i & 127;
      async_cp16(ldsBase + (u32)(b * (APITCH * 2) + c * 16),
                 gHst + (u64)(BB * HH * 2) + (u64)b * 2048 + (u64)c * 16);
    }
    async_wait0();
    __syncthreads();
    gemm_gates4<32>(acc, bufA, d1hhW, 0);

    u16 hb1[4][8];
#pragma unroll
    for (int q = 0; q < 4; ++q)
#pragma unroll
      for (int v = 0; v < 8; ++v) {
        int b = v + (hi << 3);
        int j = w * 64 + q * 16 + l15;
        float iv = sigm(acc[0][q].f[v]);
        float fv = sigm(acc[1][q].f[v]);
        float gv = tanh_f(acc[2][q].f[v]);
        float ov = sigm(acc[3][q].f[v]);
        size_t ci = (size_t)BB * HH + (size_t)b * HH + j;
        float c = fv * c_state[ci] + iv * gv;
        c_state[ci] = c;
        hb1[q][v] = f2bf(ov * tanh_f(c));
      }
    __syncthreads();
#pragma unroll
    for (int q = 0; q < 4; ++q)
#pragma unroll
      for (int v = 0; v < 8; ++v) {
        int b = v + (hi << 3);
        int j = w * 64 + q * 16 + l15;
        bufA[b * APITCH + j] = hb1[q][v];               // dec_out -> A staging
        h_state[(size_t)BB * HH + (size_t)b * HH + j] = hb1[q][v];
      }
    __syncthreads();

    // ---- q = dec_out @ attn_w^T ----
    FragC qa[4];
#pragma unroll
    for (int q = 0; q < 4; ++q)
#pragma unroll
      for (int v = 0; v < 8; ++v) qa[q].f[v] = 0.f;
    gemm_row4<32>(qa, bufA, attnW, 0, 0);
#pragma unroll
    for (int q = 0; q < 4; ++q)
#pragma unroll
      for (int v = 0; v < 8; ++v) {
        int b = v + (hi << 3);
        int n = (w * 4 + q) * 16 + l15;
        q_buf[(size_t)b * HH + n] = qa[q].f[v];
      }
    // ---- concat projection chunk 1 (K = dec_out) ----
    FragC cc[4];
#pragma unroll
    for (int q = 0; q < 4; ++q) {
      float bv = ccb[(w * 4 + q) * 16 + l15];
#pragma unroll
      for (int v = 0; v < 8; ++v) cc[q].f[v] = bv;
    }
    gemm_row4<64>(cc, bufA, ccwW, 0, 0);
    __threadfence();
    __syncthreads();

    // ---- attention (wave w handles batch b=w) ----
    {
      int b = w;
      const u16* hs = hsrc + (size_t)b * (SS * HH);
      const float* qp = q_buf + (size_t)b * HH;
      float qa0[16], qa1[16];
#pragma unroll
      for (int i = 0; i < 16; ++i) {
        int h0 = (lane << 1) + (i << 6);
        qa0[i] = qp[h0]; qa1[i] = qp[h0 + 1];
      }
      for (int s = 0; s < SS; ++s) {
        const u32* hr = (const u32*)(hs + (size_t)s * HH);
        float a = 0.f;
#pragma unroll
        for (int i = 0; i < 16; ++i) {
          u32 d = hr[lane + (i << 5)];
          a += bf2f((u16)d) * qa0[i] + bf2f((u16)(d >> 16)) * qa1[i];
        }
        a = wsum(a);
        if (lane == 0) scoreB[b * SS + s] = a;
      }
      __syncthreads();
      float s0 = scoreB[b * SS + lane], s1 = scoreB[b * SS + lane + 32];
      float m = wmaxr(fmaxf(s0, s1));
      float e0 = __expf(s0 - m), e1 = __expf(s1 - m);
      float inv = rcp_f(wsum(e0 + e1));
      scoreB[b * SS + lane] = e0 * inv;
      scoreB[b * SS + lane + 32] = e1 * inv;
      __syncthreads();
      float c0[16], c1[16];
#pragma unroll
      for (int i = 0; i < 16; ++i) { c0[i] = 0.f; c1[i] = 0.f; }
      for (int s = 0; s < SS; ++s) {
        float wv = scoreB[b * SS + s];
        const u32* hr = (const u32*)(hs + (size_t)s * HH);
#pragma unroll
        for (int i = 0; i < 16; ++i) {
          u32 d = hr[lane + (i << 5)];
          c0[i] += wv * bf2f((u16)d);
          c1[i] += wv * bf2f((u16)(d >> 16));
        }
      }
#pragma unroll
      for (int i = 0; i < 16; ++i) {
        int h0 = (lane << 1) + (i << 6);
        *(u32*)(bufA + b * APITCH + h0) = pk2(c0[i], c1[i]);   // ctx -> A staging
      }
    }
    __syncthreads();
    // ---- concat projection chunk 2 (K = ctx) ----
    gemm_row4<64>(cc, bufA, ccwW, 0, 32);
#pragma unroll
    for (int q = 0; q < 4; ++q)
#pragma unroll
      for (int v = 0; v < 8; ++v) {
        int b = v + (hi << 3);
        int n = (w * 4 + q) * 16 + l15;
        u16 hb = f2bf(tanh_f(cc[q].f[v]));
        htilde_state[(size_t)b * HH + n] = hb;
        htl_all[((size_t)b * TT + t) * HH + n] = hb;
      }
    __threadfence();
    __syncthreads();
  }
}

// ---------------- generator GEMM: logits = h_tildes @ gen_w^T + gen_b ----------------
__global__ __launch_bounds__(256) void k_gen(
    const u16* __restrict__ htl, const u32* __restrict__ gw,
    const float* __restrict__ gb, float* __restrict__ logits) {
  __shared__ __align__(16) u16 Ab[BB * GPITCH];
  int tid = threadIdx.x;
  int w = tid >> 5, lane = tid & 31, hi = lane >> 4, l15 = lane & 15;
  int r0 = blockIdx.y * 16;
  {
    u32 ldsBase = lds_addr(Ab);
    u64 src = (u64)(size_t)(htl + (size_t)r0 * HH);
    for (int i = tid; i < BB * 128; i += 256) {    // 128 x 16B chunks per row
      int b = i >> 7, c = i & 127;
      async_cp16(ldsBase + (u32)(b * (GPITCH * 2) + c * 16), src + (u64)b * 2048 + (u64)c * 16);
    }
    async_wait0();
  }
  __syncthreads();
  FragC acc[4];
  int ntBase = blockIdx.x * 32 + w * 4;
  const u32* gwW = gw + (size_t)ntBase * 32 * 256;
#pragma unroll
  for (int q = 0; q < 4; ++q)
#pragma unroll
    for (int v = 0; v < 8; ++v) acc[q].f[v] = 0.f;
  for (int kt = 0; kt < 32; ++kt) {
    Frag16 a; loadA(a, Ab, GPITCH, kt);
    const u32* pk0 = gwW + (size_t)kt * 256;
#pragma unroll
    for (int q = 0; q < 4; ++q) {
      if (ntBase + q < 2000) {
        const u32* tb = pk0 + (size_t)q * (32 * 256);
        if (kt < 31) __builtin_prefetch(tb + 256, 0, 1);
        Frag16 bfr; loadB(bfr, tb);
        acc[q].v = wmma_bf(a, bfr, acc[q].v);
      }
    }
  }
#pragma unroll
  for (int q = 0; q < 4; ++q) {
    int nt = ntBase + q;
    if (nt >= 2000) continue;
    int n = nt * 16 + l15;
    float bv = gb[n];
#pragma unroll
    for (int v = 0; v < 8; ++v) {
      int r = r0 + v + (hi << 3);
      logits[(size_t)r * VV + n] = acc[q].f[v] + bv;
    }
  }
}

// ---------------- log_softmax over V per row ----------------
__global__ __launch_bounds__(256) void k_lsm(const float* __restrict__ lg, float* __restrict__ out) {
  __shared__ float red[8];
  int r = blockIdx.x;
  const float* row = lg + (size_t)r * VV;
  int tid = threadIdx.x, lane = tid & 31, w = tid >> 5;
  float m = -3.4e38f;
  for (int i = tid; i < VV; i += 256) m = fmaxf(m, row[i]);
  m = wmaxr(m);
  if (lane == 0) red[w] = m;
  __syncthreads();
  if (w == 0) {
    float v = (lane < 8) ? red[lane] : -3.4e38f;
    v = wmaxr(v);
    if (lane == 0) red[0] = v;
  }
  __syncthreads();
  m = red[0];
  __syncthreads();
  float s = 0.f;
  for (int i = tid; i < VV; i += 256) s += __expf(row[i] - m);
  s = wsum(s);
  if (lane == 0) red[w] = s;
  __syncthreads();
  if (w == 0) {
    float v = (lane < 8) ? red[lane] : 0.f;
    v = wsum(v);
    if (lane == 0) red[0] = v;
  }
  __syncthreads();
  float lse = m + __logf(red[0]);
  float* orow = out + (size_t)r * VV;
  for (int i = tid; i < VV; i += 256) orow[i] = row[i] - lse;
}

extern "C" void kernel_launch(void* const* d_in, const int* in_sizes, int n_in,
                              void* d_out, int out_size, void* d_ws, size_t ws_size,
                              hipStream_t stream) {
  (void)in_sizes; (void)n_in; (void)out_size; (void)ws_size;
  // Assumed input order = setup_inputs dict insertion order (src, tgt, then params
  // in insertion order with enc layers as fwd,bwd tuples of (w_ih,w_hh,b_ih,b_hh)).
  const int*   src  = (const int*)d_in[0];
  const int*   tgt  = (const int*)d_in[1];
  const float* embS = (const float*)d_in[2];
  const float* embD = (const float*)d_in[3];
  const float* e0f_ih = (const float*)d_in[4];
  const float* e0f_hh = (const float*)d_in[5];
  const float* e0f_bi = (const float*)d_in[6];
  const float* e0f_bh = (const float*)d_in[7];
  const float* e0b_ih = (const float*)d_in[8];
  const float* e0b_hh = (const float*)d_in[9];
  const float* e0b_bi = (const float*)d_in[10];
  const float* e0b_bh = (const float*)d_in[11];
  const float* e1f_ih = (const float*)d_in[12];
  const float* e1f_hh = (const float*)d_in[13];
  const float* e1f_bi = (const float*)d_in[14];
  const float* e1f_bh = (const float*)d_in[15];
  const float* e1b_ih = (const float*)d_in[16];
  const float* e1b_hh = (const float*)d_in[17];
  const float* e1b_bi = (const float*)d_in[18];
  const float* e1b_bh = (const float*)d_in[19];
  const float* d0_ih = (const float*)d_in[20];
  const float* d0_hh = (const float*)d_in[21];
  const float* d0_bi = (const float*)d_in[22];
  const float* d0_bh = (const float*)d_in[23];
  const float* d1_ih = (const float*)d_in[24];
  const float* d1_hh = (const float*)d_in[25];
  const float* d1_bi = (const float*)d_in[26];
  const float* d1_bh = (const float*)d_in[27];
  const float* attnW = (const float*)d_in[28];
  const float* ccW   = (const float*)d_in[29];
  const float* ccB   = (const float*)d_in[30];
  const float* genW  = (const float*)d_in[31];
  const float* genB  = (const float*)d_in[32];

  char* ws = (char*)d_ws;
  size_t cur = 0;
  auto A = [&](size_t bytes) -> void* {
    void* p = ws + cur;
    cur += (bytes + 255) & ~(size_t)255;
    return p;
  };
  const size_t TB = 1024;  // bytes per packed tile
  u32* pkE0Fih = (u32*)A(128 * 16 * TB);
  u32* pkE0Fhh = (u32*)A(128 * 16 * TB);
  u32* pkE0Bih = (u32*)A(128 * 16 * TB);
  u32* pkE0Bhh = (u32*)A(128 * 16 * TB);
  u32* pkE1Fih = (u32*)A(128 * 32 * TB);
  u32* pkE1Fhh = (u32*)A(128 * 16 * TB);
  u32* pkE1Bih = (u32*)A(128 * 32 * TB);
  u32* pkE1Bhh = (u32*)A(128 * 16 * TB);
  u32* pkD0ih  = (u32*)A(256 * 48 * TB);
  u32* pkD0hh  = (u32*)A(256 * 32 * TB);
  u32* pkD1ih  = (u32*)A(256 * 32 * TB);
  u32* pkD1hh  = (u32*)A(256 * 32 * TB);
  u32* pkAttn  = (u32*)A(64 * 32 * TB);
  u32* pkCc    = (u32*)A(64 * 64 * TB);
  u32* pkGen   = (u32*)A((size_t)2000 * 32 * TB);
  float* bE0F = (float*)A(2048 * 4);
  float* bE0B = (float*)A(2048 * 4);
  float* bE1F = (float*)A(2048 * 4);
  float* bE1B = (float*)A(2048 * 4);
  float* bD0  = (float*)A(4096 * 4);
  float* bD1  = (float*)A(4096 * 4);
  u16* xs0    = (u16*)A((size_t)SS * BB * EE * 2);
  u16* xs1    = (u16*)A((size_t)SS * BB * HH * 2);
  u16* hsrc   = (u16*)A((size_t)BB * SS * HH * 2);
  u16* embt   = (u16*)A((size_t)TT * BB * EE * 2);
  float* hc0  = (float*)A((size_t)2 * 2 * BB * HDD * 4);
  float* hc1  = (float*)A((size_t)2 * 2 * BB * HDD * 4);
  u16* hstate = (u16*)A((size_t)2 * BB * HH * 2);
  float* cstate = (float*)A((size_t)2 * BB * HH * 4);
  u16* htld   = (u16*)A((size_t)BB * HH * 2);
  float* qbuf = (float*)A((size_t)BB * HH * 4);
  u16* htlAll = (u16*)A((size_t)BB * TT * HH * 2);
  float* logits = (float*)A((size_t)BB * TT * VV * 4);

  k_pack<<<128 * 16, 32, 0, stream>>>(e0f_ih, pkE0Fih, 2048, 512);
  k_pack<<<128 * 16, 32, 0, stream>>>(e0f_hh, pkE0Fhh, 2048, 512);
  k_pack<<<128 * 16, 32, 0, stream>>>(e0b_ih, pkE0Bih, 2048, 512);
  k_pack<<<128 * 16, 32, 0, stream>>>(e0b_hh, pkE0Bhh, 2048, 512);
  k_pack<<<128 * 32, 32, 0, stream>>>(e1f_ih, pkE1Fih, 2048, 1024);
  k_pack<<<128 * 16, 32, 0, stream>>>(e1f_hh, pkE1Fhh, 2048, 512);
  k_pack<<<128 * 32, 32, 0, stream>>>(e1b_ih, pkE1Bih, 2048, 1024);
  k_pack<<<128 * 16, 32, 0, stream>>>(e1b_hh, pkE1Bhh, 2048, 512);
  k_pack<<<256 * 48, 32, 0, stream>>>(d0_ih, pkD0ih, 4096, 1536);
  k_pack<<<256 * 32, 32, 0, stream>>>(d0_hh, pkD0hh, 4096, 1024);
  k_pack<<<256 * 32, 32, 0, stream>>>(d1_ih, pkD1ih, 4096, 1024);
  k_pack<<<256 * 32, 32, 0, stream>>>(d1_hh, pkD1hh, 4096, 1024);
  k_pack<<<64 * 32, 32, 0, stream>>>(attnW, pkAttn, 1024, 1024);
  k_pack<<<64 * 64, 32, 0, stream>>>(ccW, pkCc, 1024, 2048);
  k_pack<<<2000 * 32, 32, 0, stream>>>(genW, pkGen, 32000, 1024);
  k_bias<<<8, 256, 0, stream>>>(e0f_bi, e0f_bh, bE0F, 2048);
  k_bias<<<8, 256, 0, stream>>>(e0b_bi, e0b_bh, bE0B, 2048);
  k_bias<<<8, 256, 0, stream>>>(e1f_bi, e1f_bh, bE1F, 2048);
  k_bias<<<8, 256, 0, stream>>>(e1b_bi, e1b_bh, bE1B, 2048);
  k_bias<<<16, 256, 0, stream>>>(d0_bi, d0_bh, bD0, 4096);
  k_bias<<<16, 256, 0, stream>>>(d1_bi, d1_bh, bD1, 4096);

  k_gather<<<SS * BB, 256, 0, stream>>>(embS, src, xs0, SS);
  k_gather<<<TT * BB, 256, 0, stream>>>(embD, tgt, embt, TT);

  k_enc<<<2, 512, 0, stream>>>(xs0, 512, pkE0Fih, pkE0Fhh, bE0F, pkE0Bih, pkE0Bhh, bE0B,
                               xs1, 0, hc0);
  k_enc<<<2, 512, 0, stream>>>(xs1, 1024, pkE1Fih, pkE1Fhh, bE1F, pkE1Bih, pkE1Bhh, bE1B,
                               hsrc, 1, hc1);
  k_decinit<<<128, 256, 0, stream>>>(hc0, hc1, hstate, cstate, htld);
  k_dec<<<1, 512, 0, stream>>>(embt, hsrc, hstate, cstate, htld, qbuf, htlAll,
                               pkD0ih, pkD0hh, bD0, pkD1ih, pkD1hh, bD1,
                               pkAttn, pkCc, ccB);
  k_gen<<<dim3(63, 64), 256, 0, stream>>>(htlAll, pkGen, genB, logits);
  k_lsm<<<BB * TT, 256, 0, stream>>>(logits, (float*)d_out);
}
